// Gate_41609643163843
// MI455X (gfx1250) — compile-verified, matrix operand order checked by reading, not run on hardware
//
#include <hip/hip_runtime.h>
#include <hip/hip_bf16.h>

// ---- types for WMMA ----
typedef __attribute__((ext_vector_type(16))) __bf16 v16bf;
typedef __attribute__((ext_vector_type(2)))  __bf16 v2bf;
typedef __attribute__((ext_vector_type(8)))  float  v8f;
typedef __attribute__((ext_vector_type(4)))  float  f4v;
typedef __attribute__((ext_vector_type(4)))  unsigned int u4v;

#define T_TOK 8192
#define D_DIM 7168
#define E_EXP 256
#define BM    32      // tokens per workgroup
#define BK    128     // K panel staged in LDS per round
#define KSTEP 32      // K per WMMA

union FragBF { v16bf v; u4v q[2]; };

// pack two fp32 -> packed bf16 dword via native __bf16 casts
// (lets clang emit hw cvt, e.g. v_cvt_pk_bf16_f32, instead of integer RNE tricks)
__device__ __forceinline__ unsigned pack_bf16(float a, float b) {
    union { v2bf v; unsigned u; } r;
    r.v[0] = (__bf16)a;
    r.v[1] = (__bf16)b;
    return r.u;
}

// ---------- kernel 1: W fp32 -> bf16 into workspace (3.67 MB, L2-resident thereafter)
__global__ __launch_bounds__(256) void cvt_w_kernel(const float* __restrict__ w,
                                                    unsigned short* __restrict__ o) {
    size_t i = ((size_t)blockIdx.x * 256 + threadIdx.x) * 4;
    f4v v = *(const f4v*)(w + i);
    uint2 p;
    p.x = pack_bf16(v.x, v.y);
    p.y = pack_bf16(v.z, v.w);
    *(uint2*)(o + i) = p;
}

// ---------- kernel 2: fused gate GEMM (bf16 WMMA) + grouped top-k routing
__global__ __launch_bounds__(256) void gate_kernel(const float* __restrict__ x,
                                                   const unsigned short* __restrict__ wbf,
                                                   const float* __restrict__ bias,
                                                   float* __restrict__ outw,
                                                   int*   __restrict__ outi) {
    __shared__ __align__(16) unsigned short xs[BM * BK];   // 8 KB  x panel (bf16)
    __shared__ float scores[BM][E_EXP];                    // 32 KB sigmoid scores
    __shared__ float sel[BM][E_EXP];                       // 32 KB score+bias (masked)
    __shared__ float gsc[BM][8];
    __shared__ unsigned keepm[BM];
    __shared__ float bsh[E_EXP];

    const int tid  = threadIdx.x;
    const int wave = tid >> 5;
    const int lane = tid & 31;
    const int t0   = blockIdx.x * BM;

    if (tid < E_EXP) bsh[tid] = bias[tid];

    v8f acc[2][2];
    #pragma unroll
    for (int m = 0; m < 2; ++m)
        #pragma unroll
        for (int n = 0; n < 2; ++n)
            #pragma unroll
            for (int v = 0; v < 8; ++v) acc[m][n][v] = 0.0f;

    const int ntile0 = wave * 2;          // this wave's experts: [ntile0*16, ntile0*16+32)
    const int frow   = lane & 15;         // M (A) / N (B) row within tile
    const int fk     = (lane >> 4) * 8;   // K chunk select per ISA 16-bit layout

    // staging map: thread -> row r = tid>>3, 16 consecutive K starting at (tid&7)*16
    const int sr = tid >> 3;
    const int sc = (tid & 7) * 16;

    for (int k0 = 0; k0 < D_DIM; k0 += BK) {
        __syncthreads();   // xs reuse guard
        {
            const f4v* src = (const f4v*)(x + (size_t)(t0 + sr) * D_DIM + k0 + sc);
            uint2* dst = (uint2*)&xs[sr * BK + sc];
            #pragma unroll
            for (int i = 0; i < 4; ++i) {
                f4v v = __builtin_nontemporal_load(src + i);   // stream x, keep W in L2
                uint2 p;
                p.x = pack_bf16(v.x, v.y);
                p.y = pack_bf16(v.z, v.w);
                dst[i] = p;
            }
        }
        __syncthreads();

        #pragma unroll
        for (int kk = 0; kk < BK; kk += KSTEP) {
            FragBF a0, a1, b0, b1;
            // A fragments from LDS (tokens mt*16+frow), K chunks fk and fk+16
            a0.q[0] = *(const u4v*)(&xs[(frow     ) * BK + kk + fk]);
            a0.q[1] = *(const u4v*)(&xs[(frow     ) * BK + kk + fk + 16]);
            a1.q[0] = *(const u4v*)(&xs[(16 + frow) * BK + kk + fk]);
            a1.q[1] = *(const u4v*)(&xs[(16 + frow) * BK + kk + fk + 16]);
            // B fragments straight from L2-resident bf16 W (row = expert)
            const unsigned short* wb0 =
                wbf + (size_t)(ntile0 * 16 + frow) * D_DIM + k0 + kk + fk;
            b0.q[0] = *(const u4v*)(wb0);
            b0.q[1] = *(const u4v*)(wb0 + 16);
            const unsigned short* wb1 = wb0 + (size_t)16 * D_DIM;
            b1.q[0] = *(const u4v*)(wb1);
            b1.q[1] = *(const u4v*)(wb1 + 16);

            acc[0][0] = __builtin_amdgcn_wmma_f32_16x16x32_bf16(false, a0.v, false, b0.v,
                            (short)0, acc[0][0], false, false);
            acc[0][1] = __builtin_amdgcn_wmma_f32_16x16x32_bf16(false, a0.v, false, b1.v,
                            (short)0, acc[0][1], false, false);
            acc[1][0] = __builtin_amdgcn_wmma_f32_16x16x32_bf16(false, a1.v, false, b0.v,
                            (short)0, acc[1][0], false, false);
            acc[1][1] = __builtin_amdgcn_wmma_f32_16x16x32_bf16(false, a1.v, false, b1.v,
                            (short)0, acc[1][1], false, false);
        }
    }

    // ---- epilogue: sigmoid -> LDS scores (C layout: lane<16 -> M=v, lane>=16 -> M=v+8)
    __syncthreads();
    #pragma unroll
    for (int mt = 0; mt < 2; ++mt) {
        #pragma unroll
        for (int nt = 0; nt < 2; ++nt) {
            int n = (ntile0 + nt) * 16 + (lane & 15);
            int mb = mt * 16 + ((lane >> 4) * 8);
            #pragma unroll
            for (int v = 0; v < 8; ++v) {
                float logit = acc[mt][nt][v];
                scores[mb + v][n] = 1.0f / (1.0f + __expf(-logit));
            }
        }
    }
    __syncthreads();

    // stage A: per (token, group): bias-added scores + top-2 sum (256 threads)
    {
        int tt = tid >> 3, g = tid & 7;
        float top1 = -1e30f, top2 = -1e30f;
        for (int j = 0; j < 32; ++j) {
            int e = g * 32 + j;
            float v = scores[tt][e] + bsh[e];
            sel[tt][e] = v;
            if (v > top1) { top2 = top1; top1 = v; }
            else if (v > top2) { top2 = v; }
        }
        gsc[tt][g] = top1 + top2;
    }
    __syncthreads();

    // stage B: top-4 groups per token (32 threads)
    if (tid < BM) {
        unsigned mask = 0;
        for (int i = 0; i < 4; ++i) {
            float best = -1e30f; int bg = 0;
            for (int g = 0; g < 8; ++g) {
                if (mask & (1u << g)) continue;
                float v = gsc[tid][g];
                if (v > best) { best = v; bg = g; }
            }
            mask |= 1u << bg;
        }
        keepm[tid] = mask;
    }
    __syncthreads();

    // stage C: mask dropped groups to -inf (256 threads)
    {
        int tt = tid >> 3, g = tid & 7;
        if (!(keepm[tt] & (1u << g))) {
            for (int j = 0; j < 32; ++j) sel[tt][g * 32 + j] = -__builtin_inff();
        }
    }
    __syncthreads();

    // stage D: top-8 experts, normalize, write (32 threads)
    if (tid < BM) {
        int t = tid;
        int   idx[8];
        float w[8];
        float s = 0.0f;
        #pragma unroll 1
        for (int i = 0; i < 8; ++i) {
            float best = -1e30f; int be = 0;
            for (int e = 0; e < E_EXP; ++e) {
                float v = sel[t][e];
                if (v > best) { best = v; be = e; }
            }
            sel[t][be] = -__builtin_inff();
            idx[i] = be;
            float scv = scores[t][be];
            w[i] = scv; s += scv;
        }
        float r = 2.5f / s;
        int gt = t0 + t;
        #pragma unroll
        for (int i = 0; i < 8; ++i) {
            outw[gt * 8 + i] = w[i] * r;
            outi[gt * 8 + i] = idx[i];
        }
    }
}

extern "C" void kernel_launch(void* const* d_in, const int* in_sizes, int n_in,
                              void* d_out, int out_size, void* d_ws, size_t ws_size,
                              hipStream_t stream) {
    const float* x    = (const float*)d_in[0];   // [8192, 7168]
    const float* wgt  = (const float*)d_in[1];   // [256, 7168]
    const float* bias = (const float*)d_in[2];   // [256]

    float* outw = (float*)d_out;                         // weights [8192,8] f32
    int*   outi = (int*)((float*)d_out + T_TOK * 8);     // indices [8192,8] i32

    unsigned short* wbf = (unsigned short*)d_ws;         // 256*7168*2 = 3.67 MB

    // 1) convert W to bf16 once
    {
        int total = E_EXP * D_DIM;           // 1,835,008 (divisible by 1024)
        int blocks = total / (256 * 4);
        cvt_w_kernel<<<blocks, 256, 0, stream>>>(wgt, wbf);
    }
    // 2) fused gate GEMM + routing
    gate_kernel<<<T_TOK / BM, 256, 0, stream>>>(x, wbf, bias, outw, outi);
}
